// DensityMessagePassing_40132174414345
// MI455X (gfx1250) — compile-verified, matrix operand order checked by reading, not run on hardware
//
#include <hip/hip_runtime.h>
#include <hip/hip_bf16.h>

typedef _Float16 half_t;
typedef __attribute__((ext_vector_type(16))) _Float16 v16h;
typedef __attribute__((ext_vector_type(8)))  _Float16 v8h;
typedef __attribute__((ext_vector_type(8)))  float    v8f;

// LDS row strides padded by 8 halves (16B) so row stride is not 0 mod 64 banks.
#define LD384 392
#define LD256 264
#define LD128 136
#define LD64  72

// ---------------------------------------------------------------------------
// Weight prepack: fp32 [K x Nn] row-major  ->  f16 WMMA B-fragments.
//   lane L, element e :  k = kt*32 + (e/8)*16 + (L/16)*8 + (e%8),  col = nt*16 + (L%16)
// ---------------------------------------------------------------------------
__global__ void prepack_kernel(const float* __restrict__ W, half_t* __restrict__ P,
                               int K, int Nn) {
  int tid = blockIdx.x * blockDim.x + threadIdx.x;
  int total = K * Nn;
  if (tid >= total) return;
  int KT = K >> 5;
  int frag = tid >> 9;
  int within = tid & 511;
  int lane = within >> 4;
  int el = within & 15;
  int nt = frag / KT;
  int kt = frag - nt * KT;
  int hi = lane >> 4;
  int col = nt * 16 + (lane & 15);
  int k = kt * 32 + (el >> 3) * 16 + hi * 8 + (el & 7);
  P[tid] = (half_t)W[(size_t)k * Nn + col];
}

__global__ void zero_kernel(float* p, int n) {
  int i = blockIdx.x * blockDim.x + threadIdx.x;
  if (i < n) p[i] = 0.f;
}

// ---------------------------------------------------------------------------
// Single-tile wave GEMM (16 rows) — used by encoder/decoder.
// ---------------------------------------------------------------------------
template <int KT, int NTILES>
__device__ __forceinline__ void wave_gemm(const half_t* At, int ldw,
                                          const half_t* __restrict__ Wp,
                                          v8f* acc, int lane) {
  const int row = lane & 15, hi = lane >> 4;
  for (int kt = 0; kt < KT; ++kt) {
    const half_t* ap = At + row * ldw + kt * 32 + hi * 8;
    v8h lo = *(const v8h*)ap;
    v8h hh = *(const v8h*)(ap + 16);
    v16h a;
#pragma unroll
    for (int i = 0; i < 8; ++i) { a[i] = lo[i]; a[i + 8] = hh[i]; }
#pragma unroll
    for (int nt = 0; nt < NTILES; ++nt) {
      v16h b = *(const v16h*)(Wp + ((size_t)(nt * KT + kt) << 9) + lane * 16);
      acc[nt] = __builtin_amdgcn_wmma_f32_16x16x32_f16(
          false, a, false, b, (short)0, acc[nt], false, false);
    }
  }
}

// ---------------------------------------------------------------------------
// Dual-tile wave GEMM: 32 rows per wave; each B fragment feeds two WMMAs
// (halves weight-cache bandwidth per FLOP vs the 16-row version).
// ---------------------------------------------------------------------------
template <int KT, int NTILES>
__device__ __forceinline__ void wave_gemm_x2(const half_t* At, int ldw,
                                             const half_t* __restrict__ Wp,
                                             v8f* acc0, v8f* acc1, int lane) {
  const int row = lane & 15, hi = lane >> 4;
  for (int kt = 0; kt < KT; ++kt) {
    const half_t* ap0 = At + row * ldw + kt * 32 + hi * 8;
    const half_t* ap1 = ap0 + 16 * ldw;
    v8h lo0 = *(const v8h*)ap0, hh0 = *(const v8h*)(ap0 + 16);
    v8h lo1 = *(const v8h*)ap1, hh1 = *(const v8h*)(ap1 + 16);
    v16h a0, a1;
#pragma unroll
    for (int i = 0; i < 8; ++i) {
      a0[i] = lo0[i]; a0[i + 8] = hh0[i];
      a1[i] = lo1[i]; a1[i + 8] = hh1[i];
    }
#pragma unroll
    for (int nt = 0; nt < NTILES; ++nt) {
      v16h b = *(const v16h*)(Wp + ((size_t)(nt * KT + kt) << 9) + lane * 16);
      acc0[nt] = __builtin_amdgcn_wmma_f32_16x16x32_f16(
          false, a0, false, b, (short)0, acc0[nt], false, false);
      acc1[nt] = __builtin_amdgcn_wmma_f32_16x16x32_f16(
          false, a1, false, b, (short)0, acc1[nt], false, false);
    }
  }
}

#define ZERO8(A) { v8f _z = {}; _Pragma("unroll") for (int _i = 0; _i < 8; ++_i) (A)[_i] = _z; }

// bias + LayerNorm(128) + SiLU epilogue for one 16-row sub-tile -> f16 LDS
// (stride LD128). Row = 16-lane half-group: reduce with xor-shuffles 1,2,4,8.
__device__ __forceinline__ void ln_silu_to_lds(v8f* acc, const float* __restrict__ b1,
                                               const float* __restrict__ g,
                                               const float* __restrict__ be,
                                               half_t* H1, int lane) {
  const int col0 = lane & 15, hi = lane >> 4;
  float gv[8], bv[8], bb[8];
#pragma unroll
  for (int nt = 0; nt < 8; ++nt) {
    int c = nt * 16 + col0;
    gv[nt] = g[c]; bv[nt] = be[c]; bb[nt] = b1[c];
  }
#pragma unroll
  for (int r = 0; r < 8; ++r) {
    float vals[8];
    float s = 0.f, q = 0.f;
#pragma unroll
    for (int nt = 0; nt < 8; ++nt) {
      float v = acc[nt][r] + bb[nt];
      vals[nt] = v; s += v; q += v * v;
    }
#pragma unroll
    for (int m = 1; m < 16; m <<= 1) {
      s += __shfl_xor(s, m, 32);
      q += __shfl_xor(q, m, 32);
    }
    float mean = s * 0.0078125f;
    float rstd = rsqrtf(q * 0.0078125f - mean * mean + 1e-5f);
    int rowm = hi * 8 + r;
#pragma unroll
    for (int nt = 0; nt < 8; ++nt) {
      float v = (vals[nt] - mean) * rstd * gv[nt] + bv[nt];
      float sg = 1.f / (1.f + __expf(-v));
      H1[rowm * LD128 + nt * 16 + col0] = (half_t)(v * sg);
    }
  }
}

// Gate path fused in registers: bias + LN + SiLU + (.)·w2 + sigmoid -> GT[row].
// No LDS round-trip, no conflicted row-major dot loads.
__device__ __forceinline__ void gate_eval(v8f* acc, const float* __restrict__ b1,
                                          const float* __restrict__ g,
                                          const float* __restrict__ be,
                                          const float* __restrict__ w2, float b2s,
                                          float* GT, int rowbase, int lane) {
  const int col0 = lane & 15, hi = lane >> 4;
  float gv[8], bv[8], bb[8], wv[8];
#pragma unroll
  for (int nt = 0; nt < 8; ++nt) {
    int c = nt * 16 + col0;
    gv[nt] = g[c]; bv[nt] = be[c]; bb[nt] = b1[c]; wv[nt] = w2[c];
  }
#pragma unroll
  for (int r = 0; r < 8; ++r) {
    float vals[8];
    float s = 0.f, q = 0.f;
#pragma unroll
    for (int nt = 0; nt < 8; ++nt) {
      float v = acc[nt][r] + bb[nt];
      vals[nt] = v; s += v; q += v * v;
    }
#pragma unroll
    for (int m = 1; m < 16; m <<= 1) {
      s += __shfl_xor(s, m, 32);
      q += __shfl_xor(q, m, 32);
    }
    float mean = s * 0.0078125f;
    float rstd = rsqrtf(q * 0.0078125f - mean * mean + 1e-5f);
    float p = 0.f;
#pragma unroll
    for (int nt = 0; nt < 8; ++nt) {
      float v = (vals[nt] - mean) * rstd * gv[nt] + bv[nt];
      float sg = 1.f / (1.f + __expf(-v));
      p += v * sg * wv[nt];
    }
#pragma unroll
    for (int m = 1; m < 16; m <<= 1) p += __shfl_xor(p, m, 32);
    if (col0 == 0)
      GT[rowbase + hi * 8 + r] = 1.f / (1.f + __expf(-(p + b2s)));
  }
}

// ---------------------------------------------------------------------------
// Encoder: out[M x 128] = f16( x[M x 64] @ W + b )   (16 rows/wave, 2 waves)
// ---------------------------------------------------------------------------
__global__ __launch_bounds__(64) void encoder_kernel(
    const float* __restrict__ x, const half_t* __restrict__ w,
    const float* __restrict__ b, half_t* __restrict__ out, int M) {
  __shared__ half_t At[2][16 * LD64];
  const int wave = threadIdx.x >> 5, lane = threadIdx.x & 31;
  const long mb = ((long)blockIdx.x * 2 + wave) * 16;
  for (int it = 0; it < 8; ++it) {
    int c = it * 32 + lane;
    int row = c >> 4, p4 = (c & 15) * 4;
    long mr = mb + row; if (mr > M - 1) mr = M - 1;
    float4 v = *(const float4*)(x + mr * 64 + p4);
    half_t* d = &At[wave][row * LD64 + p4];
    d[0] = (half_t)v.x; d[1] = (half_t)v.y; d[2] = (half_t)v.z; d[3] = (half_t)v.w;
  }
  __syncthreads();
  v8f acc[8]; ZERO8(acc);
  wave_gemm<2, 8>(At[wave], LD64, w, acc, lane);
  const int col0 = lane & 15, hi = lane >> 4;
#pragma unroll
  for (int r = 0; r < 8; ++r) {
    long mr = mb + hi * 8 + r;
    if (mr < M)
#pragma unroll
      for (int nt = 0; nt < 8; ++nt)
        out[mr * 128 + nt * 16 + col0] = (half_t)(acc[nt][r] + b[nt * 16 + col0]);
  }
}

// ---------------------------------------------------------------------------
// Decoder: out[M x 64] = f32( h[M x 128] @ W + b )
// ---------------------------------------------------------------------------
__global__ __launch_bounds__(64) void decoder_kernel(
    const half_t* __restrict__ hin, const half_t* __restrict__ w,
    const float* __restrict__ b, float* __restrict__ out, int M) {
  __shared__ half_t At[2][16 * LD128];
  const int wave = threadIdx.x >> 5, lane = threadIdx.x & 31;
  const long mb = ((long)blockIdx.x * 2 + wave) * 16;
  for (int it = 0; it < 8; ++it) {
    int c = it * 32 + lane;
    int row = c >> 4, ch = c & 15;
    long mr = mb + row; if (mr > M - 1) mr = M - 1;
    *(uint4*)&At[wave][row * LD128 + ch * 8] = *(const uint4*)(hin + mr * 128 + ch * 8);
  }
  __syncthreads();
  v8f acc[4]; { v8f z = {}; for (int i = 0; i < 4; ++i) acc[i] = z; }
  wave_gemm<4, 4>(At[wave], LD128, w, acc, lane);
  const int col0 = lane & 15, hi = lane >> 4;
#pragma unroll
  for (int r = 0; r < 8; ++r) {
    long mr = mb + hi * 8 + r;
    if (mr < M)
#pragma unroll
      for (int nt = 0; nt < 4; ++nt)
        out[mr * 64 + nt * 16 + col0] = acc[nt][r] + b[nt * 16 + col0];
  }
}

// ---------------------------------------------------------------------------
// Edge stage 1 (32 edges / wave): m = sigmoid(MLP_gate([h_dst|h_src|e])) *
// MLP_msg(...), scatter-add into aggr[dst] (f32 atomics, aggr L2-resident).
// ---------------------------------------------------------------------------
__global__ __launch_bounds__(32) void edge_msg_kernel(
    const half_t* __restrict__ h, const half_t* __restrict__ e,
    const int* __restrict__ ei,
    const half_t* __restrict__ w1g, const float* __restrict__ b1g,
    const float* __restrict__ gg, const float* __restrict__ beg,
    const float* __restrict__ w2g, const float* __restrict__ b2g,
    const half_t* __restrict__ w1m, const float* __restrict__ b1m,
    const float* __restrict__ gm, const float* __restrict__ bem,
    const half_t* __restrict__ w2m, const float* __restrict__ b2m,
    float* __restrict__ aggr, int E_) {
  __shared__ half_t At[32 * LD384];
  __shared__ half_t H1[32 * LD128];
  __shared__ float  GT[32];
  __shared__ int SRC[32], DST[32];
  const int lane = threadIdx.x & 31;
  const long eb = (long)blockIdx.x * 32;
  {
    long er = eb + lane; if (er > E_ - 1) er = E_ - 1;
    SRC[lane] = ei[er];
    DST[lane] = ei[(long)E_ + er];
  }
  __syncthreads();
  // stage msg_in = [h[dst] | h[src] | e] as f16 [32][384+pad]
#pragma unroll
  for (int s = 0; s < 3; ++s) {
    for (int it = 0; it < 16; ++it) {
      int c = it * 32 + lane;
      int row = c >> 4, ch = c & 15;
      const half_t* sp;
      if (s == 0)      sp = h + (long)DST[row] * 128;
      else if (s == 1) sp = h + (long)SRC[row] * 128;
      else { long er = eb + row; if (er > E_ - 1) er = E_ - 1; sp = e + er * 128; }
      *(uint4*)&At[row * LD384 + s * 128 + ch * 8] = *(const uint4*)(sp + ch * 8);
    }
  }
  __syncthreads();

  v8f acc0[8], acc1[8];
  // ---- gate: 384->128, LN, SiLU, dot(w2), sigmoid — fused in registers
  ZERO8(acc0); ZERO8(acc1);
  wave_gemm_x2<12, 8>(At, LD384, w1g, acc0, acc1, lane);
  {
    float b2s = b2g[0];
    gate_eval(acc0, b1g, gg, beg, w2g, b2s, GT, 0, lane);
    gate_eval(acc1, b1g, gg, beg, w2g, b2s, GT, 16, lane);
  }
  // ---- msg: 384->128, LN, SiLU
  ZERO8(acc0); ZERO8(acc1);
  wave_gemm_x2<12, 8>(At, LD384, w1m, acc0, acc1, lane);
  ln_silu_to_lds(acc0, b1m, gm, bem, H1, lane);
  ln_silu_to_lds(acc1, b1m, gm, bem, H1 + 16 * LD128, lane);
  __syncthreads();
  // ---- msg: 128->128
  ZERO8(acc0); ZERO8(acc1);
  wave_gemm_x2<4, 8>(H1, LD128, w2m, acc0, acc1, lane);
  // ---- gated scatter-add into aggr[dst]
  {
    const int col0 = lane & 15, hi = lane >> 4;
    float b2v[8];
#pragma unroll
    for (int nt = 0; nt < 8; ++nt) b2v[nt] = b2m[nt * 16 + col0];
#pragma unroll
    for (int t = 0; t < 2; ++t) {
      v8f* acc = t ? acc1 : acc0;
#pragma unroll
      for (int r = 0; r < 8; ++r) {
        int rowm = t * 16 + hi * 8 + r;
        long er = eb + rowm;
        if (er < E_) {
          float gate = GT[rowm];
          long nb = (long)DST[rowm] * 128;
#pragma unroll
          for (int nt = 0; nt < 8; ++nt) {
            float m = gate * (acc[nt][r] + b2v[nt]);
            atomicAdd(&aggr[nb + nt * 16 + col0], m);
          }
        }
      }
    }
  }
}

// ---------------------------------------------------------------------------
// Node update (32 nodes / wave): h = MLP_node([h | aggr]) + h  (in-place)
// ---------------------------------------------------------------------------
__global__ __launch_bounds__(32) void node_kernel(
    half_t* __restrict__ h, const float* __restrict__ aggr,
    const half_t* __restrict__ w1, const float* __restrict__ b1,
    const float* __restrict__ g, const float* __restrict__ be,
    const half_t* __restrict__ w2, const float* __restrict__ b2, int N_) {
  __shared__ half_t At[32 * LD256];
  __shared__ half_t H1[32 * LD128];
  const int lane = threadIdx.x & 31;
  const long nb = (long)blockIdx.x * 32;
  for (int it = 0; it < 16; ++it) {          // h part (f16 copy)
    int c = it * 32 + lane;
    int row = c >> 4, ch = c & 15;
    long nr = nb + row; if (nr > N_ - 1) nr = N_ - 1;
    *(uint4*)&At[row * LD256 + ch * 8] = *(const uint4*)(h + nr * 128 + ch * 8);
  }
  for (int it = 0; it < 32; ++it) {          // aggr part (f32 -> f16)
    int c = it * 32 + lane;
    int row = c >> 5, p4 = (c & 31) * 4;
    long nr = nb + row; if (nr > N_ - 1) nr = N_ - 1;
    float4 v = *(const float4*)(aggr + nr * 128 + p4);
    half_t* d = &At[row * LD256 + 128 + p4];
    d[0] = (half_t)v.x; d[1] = (half_t)v.y; d[2] = (half_t)v.z; d[3] = (half_t)v.w;
  }
  __syncthreads();
  v8f acc0[8], acc1[8];
  ZERO8(acc0); ZERO8(acc1);
  wave_gemm_x2<8, 8>(At, LD256, w1, acc0, acc1, lane);
  ln_silu_to_lds(acc0, b1, g, be, H1, lane);
  ln_silu_to_lds(acc1, b1, g, be, H1 + 16 * LD128, lane);
  __syncthreads();
  ZERO8(acc0); ZERO8(acc1);
  wave_gemm_x2<4, 8>(H1, LD128, w2, acc0, acc1, lane);
  const int col0 = lane & 15, hi = lane >> 4;
  float b2v[8];
#pragma unroll
  for (int nt = 0; nt < 8; ++nt) b2v[nt] = b2[nt * 16 + col0];
#pragma unroll
  for (int t = 0; t < 2; ++t) {
    v8f* acc = t ? acc1 : acc0;
#pragma unroll
    for (int r = 0; r < 8; ++r) {
      int rowm = t * 16 + hi * 8 + r;
      long nr = nb + rowm;
      if (nr < N_)
#pragma unroll
        for (int nt = 0; nt < 8; ++nt) {
          int c = nt * 16 + col0;
          float v = acc[nt][r] + b2v[nt] + (float)At[rowm * LD256 + c]; // +residual
          h[nr * 128 + c] = (half_t)v;
        }
    }
  }
}

// ---------------------------------------------------------------------------
// Edge stage 2 (32 edges / wave): e = MLP_edge([e | h[src] | h[dst]]) + e
// ---------------------------------------------------------------------------
__global__ __launch_bounds__(32) void edge_upd_kernel(
    const half_t* __restrict__ h, half_t* __restrict__ e,
    const int* __restrict__ ei,
    const half_t* __restrict__ w1, const float* __restrict__ b1,
    const float* __restrict__ g, const float* __restrict__ be,
    const half_t* __restrict__ w2, const float* __restrict__ b2, int E_) {
  __shared__ half_t At[32 * LD384];
  __shared__ half_t H1[32 * LD128];
  __shared__ int SRC[32], DST[32];
  const int lane = threadIdx.x & 31;
  const long eb = (long)blockIdx.x * 32;
  {
    long er = eb + lane; if (er > E_ - 1) er = E_ - 1;
    SRC[lane] = ei[er];
    DST[lane] = ei[(long)E_ + er];
  }
  __syncthreads();
#pragma unroll
  for (int s = 0; s < 3; ++s) {
    for (int it = 0; it < 16; ++it) {
      int c = it * 32 + lane;
      int row = c >> 4, ch = c & 15;
      const half_t* sp;
      if (s == 0) { long er = eb + row; if (er > E_ - 1) er = E_ - 1; sp = e + er * 128; }
      else if (s == 1) sp = h + (long)SRC[row] * 128;
      else             sp = h + (long)DST[row] * 128;
      *(uint4*)&At[row * LD384 + s * 128 + ch * 8] = *(const uint4*)(sp + ch * 8);
    }
  }
  __syncthreads();
  v8f acc0[8], acc1[8];
  ZERO8(acc0); ZERO8(acc1);
  wave_gemm_x2<12, 8>(At, LD384, w1, acc0, acc1, lane);
  ln_silu_to_lds(acc0, b1, g, be, H1, lane);
  ln_silu_to_lds(acc1, b1, g, be, H1 + 16 * LD128, lane);
  __syncthreads();
  ZERO8(acc0); ZERO8(acc1);
  wave_gemm_x2<4, 8>(H1, LD128, w2, acc0, acc1, lane);
  const int col0 = lane & 15, hi = lane >> 4;
  float b2v[8];
#pragma unroll
  for (int nt = 0; nt < 8; ++nt) b2v[nt] = b2[nt * 16 + col0];
#pragma unroll
  for (int t = 0; t < 2; ++t) {
    v8f* acc = t ? acc1 : acc0;
#pragma unroll
    for (int r = 0; r < 8; ++r) {
      int rowm = t * 16 + hi * 8 + r;
      long er = eb + rowm;
      if (er < E_)
#pragma unroll
        for (int nt = 0; nt < 8; ++nt) {
          int c = nt * 16 + col0;
          float v = acc[nt][r] + b2v[nt] + (float)At[rowm * LD384 + c]; // +residual e
          e[er * 128 + c] = (half_t)v;
        }
    }
  }
}

// ---------------------------------------------------------------------------
// Host side
// ---------------------------------------------------------------------------
extern "C" void kernel_launch(void* const* d_in, const int* in_sizes, int n_in,
                              void* d_out, int out_size, void* d_ws, size_t ws_size,
                              hipStream_t stream) {
  // flat input order: JAX pytree (dict keys sorted)
  // 0: edge_attr  1: edge_index(int32 [2,E])
  // 2,3: edge_dec b,w   4,5: edge_enc b,w
  // 6 + l*24 + {edge(b1,b2,be,g,w1,w2), gate(...), msg(...), node(...)}
  // 78,79: node_dec b,w   80,81: node_enc b,w   82: x
  const float* edge_attr = (const float*)d_in[0];
  const int*   edge_index = (const int*)d_in[1];
  const float* edec_b = (const float*)d_in[2];
  const float* edec_w = (const float*)d_in[3];
  const float* eenc_b = (const float*)d_in[4];
  const float* eenc_w = (const float*)d_in[5];
  const float* LP[3][24];
  for (int l = 0; l < 3; ++l)
    for (int j = 0; j < 24; ++j)
      LP[l][j] = (const float*)d_in[6 + l * 24 + j];
  const float* ndec_b = (const float*)d_in[78];
  const float* ndec_w = (const float*)d_in[79];
  const float* nenc_b = (const float*)d_in[80];
  const float* nenc_w = (const float*)d_in[81];
  const float* x = (const float*)d_in[82];

  int E_ = (n_in > 0) ? in_sizes[0] / 64 : 800000;
  int N_ = (n_in > 82) ? in_sizes[82] / 64 : 50000;
  if (E_ <= 0) E_ = 800000;
  if (N_ <= 0) N_ = 50000;

  // ---- workspace carve ----
  char* ws = (char*)d_ws;
  auto carve = [&](size_t bytes) -> char* {
    char* p = ws;
    ws += (bytes + 255) & ~(size_t)255;
    return p;
  };
  half_t* hbuf = (half_t*)carve((size_t)N_ * 128 * 2);
  half_t* ebuf = (half_t*)carve((size_t)E_ * 128 * 2);
  float*  aggr = (float*)carve((size_t)N_ * 128 * 4);
  half_t* p_nenc = (half_t*)carve(64 * 128 * 2);
  half_t* p_eenc = (half_t*)carve(64 * 128 * 2);
  half_t* p_ndec = (half_t*)carve(128 * 64 * 2);
  half_t* p_edec = (half_t*)carve(128 * 64 * 2);
  half_t *p_gw1[3], *p_mw1[3], *p_mw2[3], *p_nw1[3], *p_nw2[3], *p_ew1[3], *p_ew2[3];
  for (int l = 0; l < 3; ++l) {
    p_ew1[l] = (half_t*)carve(384 * 128 * 2);
    p_ew2[l] = (half_t*)carve(128 * 128 * 2);
    p_gw1[l] = (half_t*)carve(384 * 128 * 2);
    p_mw1[l] = (half_t*)carve(384 * 128 * 2);
    p_mw2[l] = (half_t*)carve(128 * 128 * 2);
    p_nw1[l] = (half_t*)carve(256 * 128 * 2);
    p_nw2[l] = (half_t*)carve(128 * 128 * 2);
  }

  // ---- prepack weight matrices into WMMA fragment layout ----
  auto pack = [&](const float* W, half_t* P, int K, int Nn) {
    int tot = K * Nn;
    prepack_kernel<<<(tot + 255) / 256, 256, 0, stream>>>(W, P, K, Nn);
  };
  pack(nenc_w, p_nenc, 64, 128);
  pack(eenc_w, p_eenc, 64, 128);
  pack(ndec_w, p_ndec, 128, 64);
  pack(edec_w, p_edec, 128, 64);
  for (int l = 0; l < 3; ++l) {
    pack(LP[l][4],  p_ew1[l], 384, 128);   // edge.w1
    pack(LP[l][5],  p_ew2[l], 128, 128);   // edge.w2
    pack(LP[l][10], p_gw1[l], 384, 128);   // gate.w1 (gate.w2 stays fp32)
    pack(LP[l][16], p_mw1[l], 384, 128);   // msg.w1
    pack(LP[l][17], p_mw2[l], 128, 128);   // msg.w2
    pack(LP[l][22], p_nw1[l], 256, 128);   // node.w1
    pack(LP[l][23], p_nw2[l], 128, 128);   // node.w2
  }

  const int blocksN16 = ((N_ + 15) / 16 + 1) / 2;
  const int blocksE16 = ((E_ + 15) / 16 + 1) / 2;
  const int blocksN32 = (N_ + 31) / 32;
  const int blocksE32 = (E_ + 31) / 32;

  // ---- encoders ----
  encoder_kernel<<<blocksN16, 64, 0, stream>>>(x, p_nenc, nenc_b, hbuf, N_);
  encoder_kernel<<<blocksE16, 64, 0, stream>>>(edge_attr, p_eenc, eenc_b, ebuf, E_);

  // ---- 3 message-passing layers ----
  int naggr = N_ * 128;
  for (int l = 0; l < 3; ++l) {
    zero_kernel<<<(naggr + 255) / 256, 256, 0, stream>>>(aggr, naggr);
    edge_msg_kernel<<<blocksE32, 32, 0, stream>>>(
        hbuf, ebuf, edge_index,
        p_gw1[l], LP[l][6],  LP[l][9],  LP[l][8],  LP[l][11], LP[l][7],
        p_mw1[l], LP[l][12], LP[l][15], LP[l][14], p_mw2[l],  LP[l][13],
        aggr, E_);
    node_kernel<<<blocksN32, 32, 0, stream>>>(
        hbuf, aggr, p_nw1[l], LP[l][18], LP[l][21], LP[l][20], p_nw2[l], LP[l][19], N_);
    edge_upd_kernel<<<blocksE32, 32, 0, stream>>>(
        hbuf, ebuf, edge_index,
        p_ew1[l], LP[l][0], LP[l][3], LP[l][2], p_ew2[l], LP[l][1], E_);
  }

  // ---- decoders (fp32 out, concatenated) ----
  decoder_kernel<<<blocksN16, 64, 0, stream>>>(hbuf, p_ndec, ndec_b, (float*)d_out, N_);
  decoder_kernel<<<blocksE16, 64, 0, stream>>>(ebuf, p_edec, edec_b,
                                               (float*)d_out + (size_t)N_ * 64, E_);
  (void)out_size; (void)ws_size;
}
